// GroupedQueryAttention_25950192402649
// MI455X (gfx1250) — matrix operand order repaired and checked by reading, not compiled, not run on hardware
//
#include <hip/hip_runtime.h>
#include <hip/hip_bf16.h>

// ---------------- problem constants ----------------
#define B_    2
#define S_    2048
#define DM    2048          // d_model
#define H_    32            // query heads
#define KVH_  8             // kv heads
#define HD_   64            // head dim
#define NQKV  3072          // H*HD + 2*KVH*HD
#define MROWS 4096          // B*S

typedef __bf16 bf16_t;
typedef __attribute__((ext_vector_type(16))) __bf16 v16bf;
typedef __attribute__((ext_vector_type(8)))  float  v8f;
typedef __attribute__((ext_vector_type(4)))  unsigned int v4u;
typedef __attribute__((ext_vector_type(8)))  int v8i;
typedef __attribute__((ext_vector_type(4)))  int v4i;

union FragB {
    uint4 q[2];
    v16bf v;
};

static __device__ __forceinline__ unsigned short f2bf(float f) {
    unsigned u = __builtin_bit_cast(unsigned, f);
    u += 0x7FFFu + ((u >> 16) & 1u);          // round-to-nearest-even
    return (unsigned short)(u >> 16);
}

#if defined(__HIP_DEVICE_COMPILE__) && __has_builtin(__builtin_amdgcn_tensor_load_to_lds) && __has_builtin(__builtin_amdgcn_s_wait_tensorcnt)
#define USE_TDM 1
#else
#define USE_TDM 0
#endif

#if USE_TDM
// LDS byte offset of a __shared__ object (generic -> AS(3) -> int folds to the
// static LDS offset).
static __device__ __forceinline__ unsigned lds_addr_of(const void* p) {
    return (unsigned)(unsigned long long)
        (const __attribute__((address_space(3))) void*)p;
}

// Issue a 2D bf16 TDM tile load: tile_d0 (contiguous, elems) x tile_d1 rows,
// row stride stride_d0 elems in global memory, packed rows in LDS.
// D# layout per CDNA5 ISA ch.8 (group0 128b, group1 256b; groups 2/3 + the
// trailing group unused for a 2D tile -> zero-filled).
// This toolchain exposes the 6-arg builtin:
//   (uint32x4 g0, int32x8 g1, int32x4, int32x4, int32x8, i32 cpol)
static __device__ __forceinline__ void tdm_load_2d(unsigned lds_byte_addr,
                                                   const unsigned short* gptr,
                                                   unsigned tensor_d0,
                                                   unsigned tensor_d1,
                                                   unsigned tile_d0,
                                                   unsigned tile_d1,
                                                   unsigned stride_d0) {
    unsigned long long ga = (unsigned long long)(uintptr_t)gptr;
    v4u g0;
    g0[0] = 1u;                                        // count=1, user-mode D#
    g0[1] = lds_byte_addr;                             // lds_addr
    g0[2] = (unsigned)(ga & 0xFFFFFFFFu);              // global_addr[31:0]
    g0[3] = (unsigned)((ga >> 32) & 0x01FFFFFFu)       // global_addr[56:32]
          | (2u << 30);                                // type = 2 ("image")
    v8i g1;
    g1[0] = (int)(1u << 16);                           // data_size=1 (2 bytes)
    g1[1] = (int)((tensor_d0 & 0xFFFFu) << 16);        // tensor_dim0[15:0]
    g1[2] = (int)((tensor_d0 >> 16) |
                  ((tensor_d1 & 0xFFFFu) << 16));      // dim0 hi | dim1 lo
    g1[3] = (int)((tensor_d1 >> 16) |
                  (tile_d0 << 16));                    // dim1 hi | tile_dim0
    g1[4] = (int)(tile_d1 & 0xFFFFu);                  // tile_dim1 (tile_dim2=0)
    g1[5] = (int)stride_d0;                            // tensor_dim0_stride[31:0]
    g1[6] = 0;                                         // stride hi | dim1_stride lo
    g1[7] = 0;
    v4i z4 = {0, 0, 0, 0};
    v8i z8 = {0, 0, 0, 0, 0, 0, 0, 0};
    __builtin_amdgcn_tensor_load_to_lds(g0, g1, z4, z4, z8, 0);
}
#endif

// ---------------- kernel 1: fp32 -> bf16 convert ----------------
__global__ void cvt_f32_bf16(const float* __restrict__ src,
                             unsigned short* __restrict__ dst, int n) {
    int i = blockIdx.x * blockDim.x + threadIdx.x;
    if (i < n) dst[i] = f2bf(src[i]);
}

// ---------------- kernel 2/5: bf16 NT GEMM  C[M,N] = A[M,K] * W[N,K]^T ----------------
// WG tile 128x128, 8 waves of 64x32. Tile movement: Tensor Data Mover with LDS
// double buffering (wave0 issues DMA, TENSORcnt tracks, barrier publishes).
__global__ __launch_bounds__(256)
void gemm_bf16_nt(const unsigned short* __restrict__ A,
                  const unsigned short* __restrict__ W,
                  float* __restrict__ C, int M, int N, int K) {
    __shared__ unsigned short Abuf[2][128 * 64];
    __shared__ unsigned short Bbuf[2][128 * 64];

    const int tid  = threadIdx.x;
    const int lane = tid & 31;
    const int w    = tid >> 5;
    const int wm   = w >> 2;            // 0..1  (M groups of 64)
    const int wn   = w & 3;             // 0..3  (N groups of 32)
    const int m0   = blockIdx.x * 128;
    const int n0   = blockIdx.y * 128;

    v8f acc[4][2];
    #pragma unroll
    for (int i = 0; i < 4; i++)
        #pragma unroll
        for (int j = 0; j < 2; j++)
            acc[i][j] = (v8f){0.f, 0.f, 0.f, 0.f, 0.f, 0.f, 0.f, 0.f};

    const int nst = K / 64;             // 64-deep K stages

#if USE_TDM
    const bool issuer = (tid < 32);     // wave 0 drives the TDM
    if (issuer) {
        tdm_load_2d(lds_addr_of(&Abuf[0][0]), A + (size_t)m0 * K,
                    (unsigned)K, (unsigned)M, 64u, 128u, (unsigned)K);
        tdm_load_2d(lds_addr_of(&Bbuf[0][0]), W + (size_t)n0 * K,
                    (unsigned)K, (unsigned)N, 64u, 128u, (unsigned)K);
        if (nst > 1) {
            tdm_load_2d(lds_addr_of(&Abuf[1][0]), A + (size_t)m0 * K + 64,
                        (unsigned)K, (unsigned)M, 64u, 128u, (unsigned)K);
            tdm_load_2d(lds_addr_of(&Bbuf[1][0]), W + (size_t)n0 * K + 64,
                        (unsigned)K, (unsigned)N, 64u, 128u, (unsigned)K);
        }
    }
#else
    const int lrow = tid >> 1;          // 0..127
    const int lseg = (tid & 1) * 32;    // 0 / 32 (bf16 elements)
#endif

    for (int st = 0; st < nst; st++) {
        const int buf = st & 1;
#if USE_TDM
        if (issuer) {                   // stage st complete; st+1 may be in flight
            if (st + 1 < nst) __builtin_amdgcn_s_wait_tensorcnt(2);
            else              __builtin_amdgcn_s_wait_tensorcnt(0);
        }
        __syncthreads();                // publish tile to all waves
#else
        {
            const int k0 = st * 64;
            const uint4* ga = (const uint4*)(A + (size_t)(m0 + lrow) * K + k0 + lseg);
            const uint4* gb = (const uint4*)(W + (size_t)(n0 + lrow) * K + k0 + lseg);
            uint4 a0 = ga[0], a1 = ga[1], a2 = ga[2], a3 = ga[3];
            uint4 b0 = gb[0], b1 = gb[1], b2 = gb[2], b3 = gb[3];
            __syncthreads();
            uint4* la = (uint4*)(&Abuf[buf][0] + lrow * 64 + lseg);
            la[0] = a0; la[1] = a1; la[2] = a2; la[3] = a3;
            uint4* lb = (uint4*)(&Bbuf[buf][0] + lrow * 64 + lseg);
            lb[0] = b0; lb[1] = b1; lb[2] = b2; lb[3] = b3;
            __syncthreads();
        }
#endif
        const unsigned short* Ab = &Abuf[buf][0];
        const unsigned short* Bb = &Bbuf[buf][0];

        #pragma unroll
        for (int ks = 0; ks < 2; ks++) {
            const int kb = ks * 32;
            // Load all fragments into distinct registers first so the wmma
            // burst isn't serialized on ds-return waits.
            FragB bfr[2];
            #pragma unroll
            for (int j = 0; j < 2; j++) {
                const unsigned short* p =
                    Bb + (wn * 32 + j * 16 + (lane & 15)) * 64 + kb + (lane >> 4) * 16;
                bfr[j].q[0] = *(const uint4*)(p);
                bfr[j].q[1] = *(const uint4*)(p + 8);
            }
            FragB afr[4];
            #pragma unroll
            for (int ii = 0; ii < 4; ii++) {
                const unsigned short* base =
                    Ab + (wm * 64 + ii * 16 + (lane & 15)) * 64 + kb;
                afr[ii].q[0] = *(const uint4*)(base + (lane >> 4) * 16);
                afr[ii].q[1] = *(const uint4*)(base + (lane >> 4) * 16 + 8);
            }
            #pragma unroll
            for (int ii = 0; ii < 4; ii++)
                #pragma unroll
                for (int j = 0; j < 2; j++)
                    acc[ii][j] = __builtin_amdgcn_wmma_f32_16x16x32_bf16(
                        false, afr[ii].v, false, bfr[j].v, (short)0,
                        acc[ii][j], false, false);
        }

        __syncthreads();                // all waves done reading buf
#if USE_TDM
        if (issuer && st + 2 < nst) {   // refill this buffer with stage st+2
            const int k2 = (st + 2) * 64;
            tdm_load_2d(lds_addr_of(&Abuf[buf][0]), A + (size_t)m0 * K + k2,
                        (unsigned)K, (unsigned)M, 64u, 128u, (unsigned)K);
            tdm_load_2d(lds_addr_of(&Bbuf[buf][0]), W + (size_t)n0 * K + k2,
                        (unsigned)K, (unsigned)N, 64u, 128u, (unsigned)K);
        }
#endif
    }

    // C store: lane holds col = (lane&15), rows = 8*(lane>>4) + r
    #pragma unroll
    for (int i = 0; i < 4; i++)
        #pragma unroll
        for (int j = 0; j < 2; j++) {
            const int col  = n0 + wn * 32 + j * 16 + (lane & 15);
            const int rowb = m0 + wm * 64 + i * 16 + (lane >> 4) * 8;
            #pragma unroll
            for (int r = 0; r < 8; r++)
                C[(size_t)(rowb + r) * N + col] = acc[i][j][r];
        }
}

// ---------------- kernel 3: per-head RMSNorm + RoPE + relayout ----------------
__global__ __launch_bounds__(256)
void rope_kernel(const float* __restrict__ qkv,
                 const float* __restrict__ cosT, const float* __restrict__ sinT,
                 const float* __restrict__ qw,   const float* __restrict__ kw,
                 unsigned short* __restrict__ qb,
                 unsigned short* __restrict__ kb,
                 unsigned short* __restrict__ vb) {
    const int gw   = blockIdx.x * 8 + (threadIdx.x >> 5);
    const int lane = threadIdx.x & 31;
    const int row  = gw / 48;
    const int slot = gw % 48;
    const int b    = row / S_;
    const int s    = row % S_;
    const int d0   = lane;
    const int d1   = lane + 32;

    if (slot >= 40) {                    // V: convert + head-major relayout only
        const int h = slot - 40;
        const float* p = qkv + (size_t)row * NQKV + 2560 + h * 64;
        unsigned short* o = vb + ((size_t)(b * KVH_ + h) * S_ + s) * 64;
        o[d0] = f2bf(p[d0]);
        o[d1] = f2bf(p[d1]);
        return;
    }
    const bool isq = slot < 32;
    const int  h   = isq ? slot : slot - 32;
    const float* p = qkv + (size_t)row * NQKV + (isq ? h * 64 : 2048 + h * 64);
    const float* wn = isq ? qw : kw;
    unsigned short* o = isq ? qb + ((size_t)(b * H_   + h) * S_ + s) * 64
                            : kb + ((size_t)(b * KVH_ + h) * S_ + s) * 64;

    float x1 = p[d0], x2 = p[d1];
    float ss = x1 * x1 + x2 * x2;
    #pragma unroll
    for (int m = 16; m >= 1; m >>= 1) ss += __shfl_xor(ss, m, 32);
    const float scale = rsqrtf(ss * (1.f / 64.f) + 1e-6f);

    const float y1 = x1 * scale * wn[d0];
    const float y2 = x2 * scale * wn[d1];
    const float c0 = cosT[s * 64 + d0], s0 = sinT[s * 64 + d0];
    const float c1 = cosT[s * 64 + d1], s1 = sinT[s * 64 + d1];
    o[d0] = f2bf(y1 * c0 - y2 * s0);     // rotate_half: -x2 pairs with d0
    o[d1] = f2bf(y2 * c1 + y1 * s1);     //              +x1 pairs with d1
}

// ---------------- kernel 4: causal flash attention (GQA, bf16 WMMA) ----------------
__global__ __launch_bounds__(128)
void attn_kernel(const unsigned short* __restrict__ Q,
                 const unsigned short* __restrict__ Kc,
                 const unsigned short* __restrict__ Vc,
                 unsigned short* __restrict__ O) {
    __shared__ unsigned short Kt[32 * 64];        // [key][hd]
    __shared__ unsigned short Vt[64 * 32];        // [hd][key]  (transposed)
    __shared__ unsigned short Pb[4 * 16 * 32];    // per-wave P scratch

    const int tid  = threadIdx.x;
    const int lane = tid & 31;
    const int w    = tid >> 5;
    const int qblk = blockIdx.x;
    const int h    = blockIdx.y;
    const int b    = blockIdx.z;
    const int kvh  = h >> 2;                       // N_REP = 4
    const int q0   = qblk * 64 + w * 16;
    const size_t qrow  = (size_t)(b * H_ + h) * S_ + q0;
    const size_t krow0 = (size_t)(b * KVH_ + kvh) * S_;

    // Q fragments (16 rows x 64 hd = two A fragments), loaded once
    FragB qf[2];
    {
        const int m = lane & 15, kh = lane >> 4;
        const unsigned short* base = Q + (qrow + m) * 64;
        #pragma unroll
        for (int f = 0; f < 2; f++) {
            qf[f].q[0] = *(const uint4*)(base + f * 32 + kh * 8);
            qf[f].q[1] = *(const uint4*)(base + f * 32 + 16 + kh * 8);
        }
    }

    v8f oacc[4];
    #pragma unroll
    for (int jh = 0; jh < 4; jh++)
        oacc[jh] = (v8f){0.f, 0.f, 0.f, 0.f, 0.f, 0.f, 0.f, 0.f};
    float mrun[8], srun[8];
    #pragma unroll
    for (int r = 0; r < 8; r++) { mrun[r] = -1e30f; srun[r] = 0.f; }

    const int nkb  = qblk * 2 + 2;                 // causal key-block count
    const int lrow = tid >> 2;                     // 0..31 key row
    const int lseg = (tid & 3) * 16;               // hd segment

    for (int kb = 0; kb < nkb; kb++) {
        // cooperative K/V tile load
        const unsigned short* kg = Kc + (krow0 + kb * 32 + lrow) * 64 + lseg;
        const unsigned short* vg = Vc + (krow0 + kb * 32 + lrow) * 64 + lseg;
        uint4 k0v = ((const uint4*)kg)[0], k1v = ((const uint4*)kg)[1];
        uint4 v0v = ((const uint4*)vg)[0], v1v = ((const uint4*)vg)[1];
        __syncthreads();
        *(uint4*)(Kt + lrow * 64 + lseg)     = k0v;
        *(uint4*)(Kt + lrow * 64 + lseg + 8) = k1v;
        union { uint4 q; unsigned short u[8]; } t0, t1;
        t0.q = v0v; t1.q = v1v;
        #pragma unroll
        for (int e = 0; e < 8; e++) {              // transpose V into [hd][key]
            Vt[(lseg + e) * 32 + lrow]     = t0.u[e];
            Vt[(lseg + 8 + e) * 32 + lrow] = t1.u[e];
        }
        __syncthreads();

        // scores S = Q K^T : 2 key-subtiles x 2 k-steps = 4 wmma
        v8f sacc[2];
        #pragma unroll
        for (int jt = 0; jt < 2; jt++) {
            sacc[jt] = (v8f){0.f, 0.f, 0.f, 0.f, 0.f, 0.f, 0.f, 0.f};
            const int key = jt * 16 + (lane & 15);
            const unsigned short* kbase = Kt + key * 64 + (lane >> 4) * 8;
            #pragma unroll
            for (int ks = 0; ks < 2; ks++) {
                FragB bk;
                bk.q[0] = *(const uint4*)(kbase + ks * 32);
                bk.q[1] = *(const uint4*)(kbase + ks * 32 + 16);
                sacc[jt] = __builtin_amdgcn_wmma_f32_16x16x32_bf16(
                    false, qf[ks].v, false, bk.v, (short)0, sacc[jt], false, false);
            }
        }

        // online softmax (row stats via 16-lane shfl_xor reductions)
        const int key0 = kb * 32 + (lane & 15);
        const int key1 = key0 + 16;
        const int rowq = q0 + 8 * (lane >> 4);
        float p0[8], p1[8];
        #pragma unroll
        for (int r = 0; r < 8; r++) {
            float v0 = sacc[0][r] * 0.125f;
            float v1 = sacc[1][r] * 0.125f;
            const int qr = rowq + r;
            if (key0 > qr) v0 = -1e30f;
            if (key1 > qr) v1 = -1e30f;
            float mx = fmaxf(v0, v1);
            #pragma unroll
            for (int msk = 8; msk >= 1; msk >>= 1)
                mx = fmaxf(mx, __shfl_xor(mx, msk, 32));
            const float mnew = fmaxf(mrun[r], mx);
            const float corr = __expf(mrun[r] - mnew);
            const float e0 = __expf(v0 - mnew);
            const float e1 = __expf(v1 - mnew);
            float es = e0 + e1;
            #pragma unroll
            for (int msk = 8; msk >= 1; msk >>= 1)
                es += __shfl_xor(es, msk, 32);
            srun[r] = srun[r] * corr + es;
            mrun[r] = mnew;
            p0[r] = e0; p1[r] = e1;
            #pragma unroll
            for (int jh = 0; jh < 4; jh++) oacc[jh][r] *= corr;
        }

        // re-layout P (C layout -> A fragment) through per-wave LDS scratch
        unsigned short* pb = Pb + w * 512;
        #pragma unroll
        for (int r = 0; r < 8; r++) {
            const int prow = r + 8 * (lane >> 4);
            pb[prow * 32 + (lane & 15)]      = f2bf(p0[r]);
            pb[prow * 32 + 16 + (lane & 15)] = f2bf(p1[r]);
        }
        FragB pf;
        {
            const int m = lane & 15, kh = lane >> 4;
            pf.q[0] = *(const uint4*)(pb + m * 32 + kh * 8);
            pf.q[1] = *(const uint4*)(pb + m * 32 + 16 + kh * 8);
        }

        // O += P V : 4 hd-subtiles, K = 32 keys -> 4 wmma
        #pragma unroll
        for (int jh = 0; jh < 4; jh++) {
            const unsigned short* vbase = Vt + (jh * 16 + (lane & 15)) * 32 + (lane >> 4) * 8;
            FragB bv;
            bv.q[0] = *(const uint4*)(vbase);
            bv.q[1] = *(const uint4*)(vbase + 16);
            oacc[jh] = __builtin_amdgcn_wmma_f32_16x16x32_bf16(
                false, pf.v, false, bv.v, (short)0, oacc[jh], false, false);
        }
    }

    // normalize and store bf16 in [b, s, h*64+hd] layout for the output GEMM
    #pragma unroll
    for (int r = 0; r < 8; r++) {
        const float inv = 1.f / srun[r];
        const int rr = r + 8 * (lane >> 4);
        #pragma unroll
        for (int jh = 0; jh < 4; jh++)
            O[(size_t)(b * S_ + q0 + rr) * (size_t)DM + h * 64 + jh * 16 + (lane & 15)] =
                f2bf(oacc[jh][r] * inv);
    }
}

// ---------------- launcher ----------------
extern "C" void kernel_launch(void* const* d_in, const int* in_sizes, int n_in,
                              void* d_out, int out_size, void* d_ws, size_t ws_size,
                              hipStream_t stream) {
    const float* x    = (const float*)d_in[0];
    const float* rc   = (const float*)d_in[1];
    const float* rs   = (const float*)d_in[2];
    const float* Wq   = (const float*)d_in[3];
    const float* Wk   = (const float*)d_in[4];
    const float* Wv   = (const float*)d_in[5];
    const float* Wo   = (const float*)d_in[6];
    const float* qnw  = (const float*)d_in[7];
    const float* knw  = (const float*)d_in[8];
    float* out = (float*)d_out;

    char* ws = (char*)d_ws;
    unsigned short* xb    = (unsigned short*)ws; ws += (size_t)MROWS * DM * 2;
    unsigned short* wqkv  = (unsigned short*)ws; ws += (size_t)NQKV * DM * 2;
    unsigned short* wob   = (unsigned short*)ws; ws += (size_t)DM * DM * 2;
    float*          qkvf  = (float*)ws;          ws += (size_t)MROWS * NQKV * 4;
    unsigned short* qb    = (unsigned short*)ws; ws += (size_t)B_ * H_   * S_ * HD_ * 2;
    unsigned short* kbuf  = (unsigned short*)ws; ws += (size_t)B_ * KVH_ * S_ * HD_ * 2;
    unsigned short* vbuf  = (unsigned short*)ws; ws += (size_t)B_ * KVH_ * S_ * HD_ * 2;
    unsigned short* attnb = (unsigned short*)ws; ws += (size_t)MROWS * DM * 2;

    // 1) convert activations + weights to bf16 (Wq/Wk/Wv concatenated -> single QKV GEMM)
    int n;
    n = MROWS * DM;
    cvt_f32_bf16<<<(n + 255) / 256, 256, 0, stream>>>(x, xb, n);
    n = H_ * HD_ * DM;
    cvt_f32_bf16<<<(n + 255) / 256, 256, 0, stream>>>(Wq, wqkv, n);
    n = KVH_ * HD_ * DM;
    cvt_f32_bf16<<<(n + 255) / 256, 256, 0, stream>>>(Wk, wqkv + (size_t)2048 * 2048, n);
    cvt_f32_bf16<<<(n + 255) / 256, 256, 0, stream>>>(Wv, wqkv + (size_t)2560 * 2048, n);
    n = DM * DM;
    cvt_f32_bf16<<<(n + 255) / 256, 256, 0, stream>>>(Wo, wob, n);

    // 2) fused QKV projection: [4096,2048] x [3072,2048]^T -> fp32 [4096,3072]
    {
        dim3 g(MROWS / 128, NQKV / 128);
        gemm_bf16_nt<<<g, 256, 0, stream>>>(xb, wqkv, qkvf, MROWS, NQKV, DM);
    }

    // 3) RMSNorm + RoPE + head-major bf16 relayout
    rope_kernel<<<(MROWS * 48) / 8, 256, 0, stream>>>(qkvf, rc, rs, qnw, knw,
                                                      qb, kbuf, vbuf);

    // 4) causal flash attention
    {
        dim3 g(S_ / 64, H_, B_);
        attn_kernel<<<g, 128, 0, stream>>>(qb, kbuf, vbuf, attnb);
    }

    // 5) output projection: [4096,2048] x [2048,2048]^T -> fp32 d_out
    {
        dim3 g(MROWS / 128, DM / 128);
        gemm_bf16_nt<<<g, 256, 0, stream>>>(attnb, wob, out, MROWS, DM, DM);
    }
}
